// GAT_19181323944126
// MI455X (gfx1250) — matrix operand order repaired and checked by reading, not compiled, not run on hardware
//
#include <hip/hip_runtime.h>
#include <hip/hip_bf16.h>
#include <math.h>

// ---------------------------------------------------------------------------
// Types for CDNA5 WMMA (wave32): v_wmma_f32_16x16x32_bf16
// ---------------------------------------------------------------------------
typedef __attribute__((ext_vector_type(16))) __bf16 v16bf;
typedef __attribute__((ext_vector_type(8)))  float  v8f;

// Split fp32 into bf16 hi + bf16 lo residual using hardware converts (RNE).
__device__ __forceinline__ void bf16_split(float v, __bf16& h, __bf16& l) {
    __bf16 bh = (__bf16)v;            // v_cvt to bf16, round-to-nearest-even
    float r = v - (float)bh;          // exact: bf16->f32 is exact
    h = bh;
    l = (__bf16)r;
}

// Order-preserving encoding of float into uint for atomicMax over signed floats
__device__ __forceinline__ unsigned enc_f32(float f) {
    unsigned u = __float_as_uint(f);
    return (u & 0x80000000u) ? ~u : (u | 0x80000000u);
}
__device__ __forceinline__ float dec_f32(unsigned e) {
    unsigned u = (e & 0x80000000u) ? (e & 0x7FFFFFFFu) : ~e;
    return __uint_as_float(u);
}

// ---------------------------------------------------------------------------
// fp32 GEMM via 3x bf16-split WMMA: C[M,Nc] = A[M,K] @ B[K,Nc]
// One wave32 computes an MT x NT group of 16x16 tiles (fragment reuse).
// K must be a multiple of 32. A rows must be 16B-aligned (K % 8 == 0).
// ---------------------------------------------------------------------------
template<int MT, int NT>
__global__ void gemm_bf16x3_wmma(const float* __restrict__ A,
                                 const float* __restrict__ B,
                                 float* __restrict__ C,
                                 int M, int K, int Nc,
                                 int groups_n, int total_groups)
{
    int wave = blockIdx.x * (blockDim.x >> 5) + (threadIdx.x >> 5);
    if (wave >= total_groups) return;             // wave-uniform exit
    int lane = threadIdx.x & 31;
    int gm = wave / groups_n;
    int gn = wave % groups_n;
    int hi = lane >> 4;                           // lane half: 0 or 1
    int lq = lane & 15;

    v8f acc[MT][NT] = {};                         // fp32 accumulators

    for (int k0 = 0; k0 < K; k0 += 32) {
        v16bf aHi[MT], aLo[MT], bHi[NT], bLo[NT];

        // --- A fragments: row M = (gm*MT+mt)*16 + lq
        //     element e holds K = k0 + (e>>3)*16 + hi*8 + (e&7)
        //     -> per lane: two contiguous 8-float runs -> 4x float4 loads
        #pragma unroll
        for (int mt = 0; mt < MT; ++mt) {
            int arow = (gm * MT + mt) * 16 + lq;
            if (arow >= M) arow = M - 1;          // clamp (rows dropped at store)
            const float* Arow = A + (size_t)arow * K + k0 + (hi << 3);
            float4 a0 = *(const float4*)(Arow);
            float4 a1 = *(const float4*)(Arow + 4);
            float4 a2 = *(const float4*)(Arow + 16);
            float4 a3 = *(const float4*)(Arow + 20);
            float av[16] = { a0.x, a0.y, a0.z, a0.w, a1.x, a1.y, a1.z, a1.w,
                             a2.x, a2.y, a2.z, a2.w, a3.x, a3.y, a3.z, a3.w };
            #pragma unroll
            for (int e = 0; e < 16; ++e) {
                __bf16 h, l; bf16_split(av[e], h, l);
                aHi[mt][e] = h; aLo[mt][e] = l;
            }
        }

        // --- B fragments: col N = (gn*NT+nt)*16 + lq ; K = k0 + (e>>3)*16 + hi*8 + (e&7)
        #pragma unroll
        for (int nt = 0; nt < NT; ++nt) {
            int bcol = (gn * NT + nt) * 16 + lq;
            if (bcol >= Nc) bcol = Nc - 1;
            const float* Bp = B + (size_t)(k0 + (hi << 3)) * Nc + bcol;
            #pragma unroll
            for (int e = 0; e < 16; ++e) {
                __bf16 h, l; bf16_split(Bp[(size_t)((e & 7) + ((e >> 3) << 4)) * Nc], h, l);
                bHi[nt][e] = h; bLo[nt][e] = l;
            }
        }

        // --- 3-term split-precision accumulate: A*B ~= Ah*Bh + Ah*Bl + Al*Bh
        #pragma unroll
        for (int mt = 0; mt < MT; ++mt)
            #pragma unroll
            for (int nt = 0; nt < NT; ++nt) {
                acc[mt][nt] = __builtin_amdgcn_wmma_f32_16x16x32_bf16(
                    false, aHi[mt], false, bHi[nt], (short)0, acc[mt][nt], false, false);
                acc[mt][nt] = __builtin_amdgcn_wmma_f32_16x16x32_bf16(
                    false, aHi[mt], false, bLo[nt], (short)0, acc[mt][nt], false, false);
                acc[mt][nt] = __builtin_amdgcn_wmma_f32_16x16x32_bf16(
                    false, aLo[mt], false, bHi[nt], (short)0, acc[mt][nt], false, false);
            }
    }

    // --- Store: element v -> row = tile_m*16 + hi*8 + v, col = tile_n*16 + lq
    #pragma unroll
    for (int mt = 0; mt < MT; ++mt) {
        int tmBase = (gm * MT + mt) * 16;
        int rbase = tmBase + (hi << 3);
        #pragma unroll
        for (int nt = 0; nt < NT; ++nt) {
            int col = (gn * NT + nt) * 16 + lq;
            if (tmBase + 16 <= M && col < Nc) {
                // full-tile fast path: no per-row guards
                float* Cp = C + (size_t)rbase * Nc + col;
                #pragma unroll
                for (int v = 0; v < 8; ++v)
                    Cp[(size_t)v * Nc] = acc[mt][nt][v];
            } else if (col < Nc) {
                #pragma unroll
                for (int v = 0; v < 8; ++v) {
                    int row = rbase + v;
                    if (row < M) C[(size_t)row * Nc + col] = acc[mt][nt][v];
                }
            }
        }
    }
}

// ---------------------------------------------------------------------------
// Utility / elementwise kernels
// ---------------------------------------------------------------------------
__global__ void fill_f32(float* __restrict__ p, float v, int n) {
    int i = blockIdx.x * blockDim.x + threadIdx.x;
    if (i < n) p[i] = v;
}

// Per-(node, head) attention scores: s = <h[n,hd,:], a[hd,:]>
__global__ void scores_kernel(const float* __restrict__ h,
                              const float* __restrict__ a_src,
                              const float* __restrict__ a_dst,
                              float* __restrict__ ssrc, float* __restrict__ sdst,
                              int N, int H, int C)
{
    int t = blockIdx.x * blockDim.x + threadIdx.x;
    if (t >= N * H) return;
    int n = t / H, hd = t % H;
    const float* hp = h + (size_t)n * H * C + (size_t)hd * C;
    const float* as = a_src + hd * C;
    const float* ad = a_dst + hd * C;
    float s1 = 0.f, s2 = 0.f;
    for (int c = 0; c < C; ++c) { float v = hp[c]; s1 += v * as[c]; s2 += v * ad[c]; }
    ssrc[t] = s1; sdst[t] = s2;
}

__device__ __forceinline__ void edge_pair(const int* __restrict__ src,
                                          const int* __restrict__ dst,
                                          int eidx, int E, int& s, int& d)
{
    if (eidx < E) { s = src[eidx]; d = dst[eidx]; }
    else          { s = eidx - E;  d = eidx - E; }   // self-loops appended
}

__global__ void edge_max_kernel(const int* __restrict__ src, const int* __restrict__ dst,
                                int E, int N, int H,
                                const float* __restrict__ ssrc,
                                const float* __restrict__ sdst,
                                unsigned* __restrict__ mEnc)
{
    int t = blockIdx.x * blockDim.x + threadIdx.x;
    int total = (E + N) * H;
    if (t >= total) return;
    int eidx = t / H, hd = t % H, s, d;
    edge_pair(src, dst, eidx, E, s, d);
    float e = ssrc[s * H + hd] + sdst[d * H + hd];
    e = e > 0.f ? e : 0.2f * e;                    // leaky_relu
    atomicMax(&mEnc[d * H + hd], enc_f32(e));
}

__global__ void edge_sum_kernel(const int* __restrict__ src, const int* __restrict__ dst,
                                int E, int N, int H,
                                const float* __restrict__ ssrc,
                                const float* __restrict__ sdst,
                                const unsigned* __restrict__ mEnc,
                                float* __restrict__ den)
{
    int t = blockIdx.x * blockDim.x + threadIdx.x;
    int total = (E + N) * H;
    if (t >= total) return;
    int eidx = t / H, hd = t % H, s, d;
    edge_pair(src, dst, eidx, E, s, d);
    int di = d * H + hd;
    float e = ssrc[s * H + hd] + sdst[di];
    e = e > 0.f ? e : 0.2f * e;
    float ex = __expf(e - dec_f32(mEnc[di]));
    atomicAdd(&den[di], ex);
}

__global__ void edge_agg_kernel(const int* __restrict__ src, const int* __restrict__ dst,
                                int E, int N, int H, int C,
                                const float* __restrict__ ssrc,
                                const float* __restrict__ sdst,
                                const unsigned* __restrict__ mEnc,
                                const float* __restrict__ den,
                                const float* __restrict__ h,
                                float* __restrict__ out)
{
    int t = blockIdx.x * blockDim.x + threadIdx.x;
    int total = (E + N) * H;
    if (t >= total) return;
    int eidx = t / H, hd = t % H, s, d;
    edge_pair(src, dst, eidx, E, s, d);
    int di = d * H + hd;
    float e = ssrc[s * H + hd] + sdst[di];
    e = e > 0.f ? e : 0.2f * e;
    float ex = __expf(e - dec_f32(mEnc[di]));
    float alpha = ex / (den[di] + 1e-16f);
    const float* hp = h + (size_t)s * H * C + (size_t)hd * C;
    float* op = out + (size_t)d * H * C + (size_t)hd * C;
    for (int c = 0; c < C; ++c)
        atomicAdd(&op[c], hp[c] * alpha);
}

__global__ void elu_bias_kernel(float* __restrict__ x, const float* __restrict__ b,
                                int stride, int n)
{
    int i = blockIdx.x * blockDim.x + threadIdx.x;
    if (i >= n) return;
    float v = x[i] + b[i % stride];
    x[i] = v > 0.f ? v : expm1f(v);
}

__global__ void add_bias_kernel(float* __restrict__ x, const float* __restrict__ b,
                                int stride, int n)
{
    int i = blockIdx.x * blockDim.x + threadIdx.x;
    if (i >= n) return;
    x[i] += b[i % stride];
}

// ---------------------------------------------------------------------------
// Host-side launcher
// ---------------------------------------------------------------------------
extern "C" void kernel_launch(void* const* d_in, const int* in_sizes, int n_in,
                              void* d_out, int out_size, void* d_ws, size_t ws_size,
                              hipStream_t stream)
{
    const float* x   = (const float*)d_in[0];
    const int*   ei  = (const int*)  d_in[1];
    const float* W1  = (const float*)d_in[2];
    const float* as1 = (const float*)d_in[3];
    const float* ad1 = (const float*)d_in[4];
    const float* b1  = (const float*)d_in[5];
    const float* W2  = (const float*)d_in[6];
    const float* as2 = (const float*)d_in[7];
    const float* ad2 = (const float*)d_in[8];
    const float* b2  = (const float*)d_in[9];
    float* out = (float*)d_out;

    const int INCH = 128, HID = 64, HEADS = 4, OUTCH = 16;
    const int H1 = HEADS * HID;               // 256
    const int N = in_sizes[0] / INCH;         // 50000
    const int E = in_sizes[1] / 2;            // 800000
    const int* srcA = ei;
    const int* dstA = ei + E;

    // ---- workspace layout (floats) ----
    float*    ws    = (float*)d_ws;
    float*    h1    = ws;                                    // N*256
    float*    out1  = h1    + (size_t)N * H1;                // N*256
    float*    h2    = out1  + (size_t)N * H1;                // N*16
    float*    ssrc1 = h2    + (size_t)N * OUTCH;             // N*4
    float*    sdst1 = ssrc1 + (size_t)N * HEADS;             // N*4
    unsigned* m1    = (unsigned*)(sdst1 + (size_t)N * HEADS);// N*4
    float*    den1  = (float*)(m1 + (size_t)N * HEADS);      // N*4
    float*    ssrc2 = den1  + (size_t)N * HEADS;             // N
    float*    sdst2 = ssrc2 + N;                             // N
    unsigned* m2    = (unsigned*)(sdst2 + N);                // N
    float*    den2  = (float*)(m2 + N);                      // N

    const int TB = 256;
    auto blocks = [](int n, int tb) { return (n + tb - 1) / tb; };

    // ---- zero initialization (enc(-inf) > 0u, so 0u works as max identity) ----
    fill_f32<<<blocks(N * H1, TB), TB, 0, stream>>>(out1, 0.f, N * H1);
    fill_f32<<<blocks(N * HEADS, TB), TB, 0, stream>>>((float*)m1, 0.f, N * HEADS);
    fill_f32<<<blocks(N * HEADS, TB), TB, 0, stream>>>(den1, 0.f, N * HEADS);
    fill_f32<<<blocks(N, TB), TB, 0, stream>>>((float*)m2, 0.f, N);
    fill_f32<<<blocks(N, TB), TB, 0, stream>>>(den2, 0.f, N);
    fill_f32<<<blocks(N * OUTCH, TB), TB, 0, stream>>>(out, 0.f, N * OUTCH);

    const int wpb = TB / 32;                  // waves per block

    // ---- layer 1: h1 = x @ W1, 16x64 strip per wave (NT=4, A-frag reuse) ----
    {
        int groups_m = (N + 15) / 16;         // 3125
        int groups_n = (H1 / 16) / 4;         // 4
        int total = groups_m * groups_n;
        gemm_bf16x3_wmma<1, 4><<<blocks(total, wpb), TB, 0, stream>>>(
            x, W1, h1, N, INCH, H1, groups_n, total);
    }
    scores_kernel<<<blocks(N * HEADS, TB), TB, 0, stream>>>(
        h1, as1, ad1, ssrc1, sdst1, N, HEADS, HID);

    int ET1 = (E + N) * HEADS;
    edge_max_kernel<<<blocks(ET1, TB), TB, 0, stream>>>(
        srcA, dstA, E, N, HEADS, ssrc1, sdst1, m1);
    edge_sum_kernel<<<blocks(ET1, TB), TB, 0, stream>>>(
        srcA, dstA, E, N, HEADS, ssrc1, sdst1, m1, den1);
    edge_agg_kernel<<<blocks(ET1, TB), TB, 0, stream>>>(
        srcA, dstA, E, N, HEADS, HID, ssrc1, sdst1, m1, den1, h1, out1);

    elu_bias_kernel<<<blocks(N * H1, TB), TB, 0, stream>>>(out1, b1, H1, N * H1);

    // ---- layer 2: h2 = elu_out @ W2, 64x16 strip per wave (MT=4, B-frag reuse) ----
    {
        int tiles_m = (N + 15) / 16;          // 3125
        int groups_m = (tiles_m + 3) / 4;     // 782 (last group partially guarded)
        int groups_n = OUTCH / 16;            // 1
        int total = groups_m * groups_n;
        gemm_bf16x3_wmma<4, 1><<<blocks(total, wpb), TB, 0, stream>>>(
            out1, W2, h2, N, H1, OUTCH, groups_n, total);
    }
    scores_kernel<<<blocks(N, TB), TB, 0, stream>>>(
        h2, as2, ad2, ssrc2, sdst2, N, 1, OUTCH);

    int ET2 = (E + N);
    edge_max_kernel<<<blocks(ET2, TB), TB, 0, stream>>>(
        srcA, dstA, E, N, 1, ssrc2, sdst2, m2);
    edge_sum_kernel<<<blocks(ET2, TB), TB, 0, stream>>>(
        srcA, dstA, E, N, 1, ssrc2, sdst2, m2, den2);
    edge_agg_kernel<<<blocks(ET2, TB), TB, 0, stream>>>(
        srcA, dstA, E, N, 1, OUTCH, ssrc2, sdst2, m2, den2, h2, out);

    add_bias_kernel<<<blocks(N * OUTCH, TB), TB, 0, stream>>>(out, b2, OUTCH, N * OUTCH);
}
